// SAM2MultiScaleBlock_3221225472682
// MI455X (gfx1250) — matrix operand order repaired and checked by reading, not compiled, not run on hardware
//
#include <hip/hip_runtime.h>
#include <math.h>

typedef unsigned short u16;
typedef __attribute__((ext_vector_type(16))) __bf16 v16bf;
typedef __attribute__((ext_vector_type(8)))  float  v8f;
typedef __attribute__((ext_vector_type(4)))  unsigned int v4u;

union Frag { v4u u[2]; v16bf v; };

#define EP_F32        0
#define EP_BF16       1
#define EP_GELU_BF16  2
#define EP_ADD_F32    3

#define DIM      112
#define DIM_OUT  224
#define HEADS    4
#define HD       56
#define MLP_DIM  896
#define NTOK     131072ull   // 8*128*128
#define NTOK2    32768ull    // 8*64*64
#define NWIN     2048
#define EPSLN    1e-6f
#define ASCALE   0.13363062095621219f  // 56^-0.5

__device__ __forceinline__ u16 f2bf(float f) {
  unsigned u = __builtin_bit_cast(unsigned, f);
  u += 0x7FFFu + ((u >> 16) & 1u);
  return (u16)(u >> 16);
}
__device__ __forceinline__ float bf2f(u16 h) {
  return __builtin_bit_cast(float, ((unsigned)h) << 16);
}
__device__ __forceinline__ float wredsum32(float v) {
#pragma unroll
  for (int m = 16; m >= 1; m >>= 1) v += __shfl_xor(v, m, 32);
  return v;
}
__device__ __forceinline__ float wredsum16(float v) {
#pragma unroll
  for (int m = 8; m >= 1; m >>= 1) v += __shfl_xor(v, m, 32);
  return v;
}
__device__ __forceinline__ float wredmax16(float v) {
#pragma unroll
  for (int m = 8; m >= 1; m >>= 1) v = fmaxf(v, __shfl_xor(v, m, 32));
  return v;
}

// ---------------------------------------------------------------------------
// Weight prep: f32 W[K][N] -> bf16 W^T[Np][Kp] (zero padded)
// ---------------------------------------------------------------------------
__global__ void convert_weight_t(const float* __restrict__ w, u16* __restrict__ wt,
                                 int K, int N, int Kp, int Np) {
  size_t i = (size_t)blockIdx.x * blockDim.x + threadIdx.x;
  if (i >= (size_t)Kp * Np) return;
  int n = (int)(i / Kp);
  int k = (int)(i % Kp);
  float v = (k < K && n < N) ? w[(size_t)k * N + n] : 0.f;
  wt[i] = f2bf(v);
}

__global__ void pad_bias(const float* __restrict__ b, float* __restrict__ bp, int N, int Np) {
  int i = blockIdx.x * blockDim.x + threadIdx.x;
  if (i < Np) bp[i] = (i < N) ? b[i] : 0.f;
}

// ---------------------------------------------------------------------------
// LayerNorm over 112 channels -> bf16 rows padded to 128. One wave per token.
// ---------------------------------------------------------------------------
__global__ __launch_bounds__(256) void ln1_to_bf16(const float* __restrict__ x,
                                                   const float* __restrict__ g,
                                                   const float* __restrict__ bt,
                                                   u16* __restrict__ out) {
  const int wave = threadIdx.x >> 5, lane = threadIdx.x & 31;
  const size_t tok = (size_t)blockIdx.x * 8 + wave;
  const float* row = x + tok * DIM;
  float v[4];
  float s = 0.f;
#pragma unroll
  for (int j = 0; j < 4; ++j) {
    int c = lane + 32 * j;
    v[j] = (c < DIM) ? row[c] : 0.f;
    s += v[j];
  }
  float mean = wredsum32(s) * (1.0f / DIM);
  float sq = 0.f;
#pragma unroll
  for (int j = 0; j < 4; ++j) {
    int c = lane + 32 * j;
    if (c < DIM) { float d = v[j] - mean; sq += d * d; }
  }
  float rstd = rsqrtf(wredsum32(sq) * (1.0f / DIM) + EPSLN);
#pragma unroll
  for (int j = 0; j < 4; ++j) {
    int c = lane + 32 * j;
    u16 o = 0;
    if (c < DIM) o = f2bf((v[j] - mean) * rstd * g[c] + bt[c]);
    out[tok * 128 + c] = o;
  }
}

// ---------------------------------------------------------------------------
// Generic bf16 WMMA GEMM: C[M][N] = A[M][Kp] * Bt[Np][Kp]^T + bias, epilogue
// Block: 256 thr = 8 waves; wave -> 32 rows x 64 cols (2x4 wmma tiles,
// B fragments reused across both row tiles: 12 b128 loads / 8 wmma / k-step).
// grid = (Np/64, M/256)
// ---------------------------------------------------------------------------
__global__ __launch_bounds__(256) void gemm_bf16_wmma(
    const u16* __restrict__ A, const u16* __restrict__ Bt, const float* __restrict__ bias,
    int Kp, int N, int ldc, float* __restrict__ Cf, u16* __restrict__ Cb,
    const float* __restrict__ addb, int mode) {
  const int wave = threadIdx.x >> 5;
  const int lane = threadIdx.x & 31;
  const int half = lane >> 4;
  const int mrow = lane & 15;
  const size_t row0 = (size_t)blockIdx.y * 256 + (size_t)wave * 32;
  const int col0 = blockIdx.x * 64;

  const u16* arow0 = A + (row0 + (size_t)mrow) * Kp;
  const u16* arow1 = arow0 + (size_t)16 * Kp;
  const u16* b0r = Bt + ((size_t)(col0 + mrow)) * Kp;
  const u16* b1r = b0r + (size_t)16 * Kp;
  const u16* b2r = b0r + (size_t)32 * Kp;
  const u16* b3r = b0r + (size_t)48 * Kp;

  v8f z = {0,0,0,0,0,0,0,0};
  v8f c00 = z, c01 = z, c02 = z, c03 = z;
  v8f c10 = z, c11 = z, c12 = z, c13 = z;

  for (int k0 = 0; k0 < Kp; k0 += 32) {
    Frag a0, a1, b0, b1, b2, b3;
    a0.u[0] = *(const v4u*)(arow0 + k0 + half * 16);
    a0.u[1] = *(const v4u*)(arow0 + k0 + half * 16 + 8);
    a1.u[0] = *(const v4u*)(arow1 + k0 + half * 16);
    a1.u[1] = *(const v4u*)(arow1 + k0 + half * 16 + 8);
    const v4u* p0 = (const v4u*)(b0r + k0 + half * 16);
    const v4u* p1 = (const v4u*)(b1r + k0 + half * 16);
    const v4u* p2 = (const v4u*)(b2r + k0 + half * 16);
    const v4u* p3 = (const v4u*)(b3r + k0 + half * 16);
    b0.u[0] = p0[0]; b0.u[1] = p0[1];
    b1.u[0] = p1[0]; b1.u[1] = p1[1];
    b2.u[0] = p2[0]; b2.u[1] = p2[1];
    b3.u[0] = p3[0]; b3.u[1] = p3[1];
    __builtin_prefetch(arow0 + k0 + 64, 0, 1);
    __builtin_prefetch(arow1 + k0 + 64, 0, 1);
    c00 = __builtin_amdgcn_wmma_f32_16x16x32_bf16(false, a0.v, false, b0.v, (short)0, c00, false, false);
    c01 = __builtin_amdgcn_wmma_f32_16x16x32_bf16(false, a0.v, false, b1.v, (short)0, c01, false, false);
    c02 = __builtin_amdgcn_wmma_f32_16x16x32_bf16(false, a0.v, false, b2.v, (short)0, c02, false, false);
    c03 = __builtin_amdgcn_wmma_f32_16x16x32_bf16(false, a0.v, false, b3.v, (short)0, c03, false, false);
    c10 = __builtin_amdgcn_wmma_f32_16x16x32_bf16(false, a1.v, false, b0.v, (short)0, c10, false, false);
    c11 = __builtin_amdgcn_wmma_f32_16x16x32_bf16(false, a1.v, false, b1.v, (short)0, c11, false, false);
    c12 = __builtin_amdgcn_wmma_f32_16x16x32_bf16(false, a1.v, false, b2.v, (short)0, c12, false, false);
    c13 = __builtin_amdgcn_wmma_f32_16x16x32_bf16(false, a1.v, false, b3.v, (short)0, c13, false, false);
  }

  v8f accs[2][4] = {{c00, c01, c02, c03}, {c10, c11, c12, c13}};
#pragma unroll
  for (int s2 = 0; s2 < 2; ++s2) {
    const size_t gm0 = row0 + (size_t)(s2 * 16) + (size_t)(8 * half);
#pragma unroll
    for (int t = 0; t < 4; ++t) {
      int gn = col0 + t * 16 + mrow;
      if (gn >= N) continue;
      float bs = bias[gn];
#pragma unroll
      for (int r = 0; r < 8; ++r) {
        float vv = accs[s2][t][r] + bs;
        size_t idx = (gm0 + r) * (size_t)ldc + gn;
        if (mode == EP_F32)            Cf[idx] = vv;
        else if (mode == EP_BF16)      Cb[idx] = f2bf(vv);
        else if (mode == EP_GELU_BF16) Cb[idx] = f2bf(0.5f * vv * (1.0f + erff(vv * 0.70710678118654752f)));
        else                           Cf[idx] = vv + addb[idx];  // EP_ADD_F32
      }
    }
  }
}

// ---------------------------------------------------------------------------
// Repack qkv (token-major bf16 [NTOK][672]) into attention-friendly layouts:
//   qp[w][h][16][64]  : pooled q * scale, channels zero padded to 64
//   kp[w][h][64][64]  : k, channel padded
//   vt[w][h][64][64]  : v transposed (channel-major), padded rows
// ---------------------------------------------------------------------------
__global__ __launch_bounds__(256) void repack_qkv(const u16* __restrict__ qkv,
                                                  u16* __restrict__ qp,
                                                  u16* __restrict__ kp,
                                                  u16* __restrict__ vt) {
  const int w = blockIdx.x;
  const int b  = w >> 8;
  const int wy = (w >> 4) & 15;
  const int wx = w & 15;
  // k, v: 4 heads * 64 tokens * 64 channels
  for (int idx = threadIdx.x; idx < 4 * 64 * 64; idx += blockDim.x) {
    int h = idx >> 12, tok = (idx >> 6) & 63, ch = idx & 63;
    u16 kv = 0, vv = 0;
    if (ch < HD) {
      int ty = tok >> 3, tx = tok & 7;
      size_t tg = (size_t)b * 16384 + (size_t)(wy * 8 + ty) * 128 + (wx * 8 + tx);
      const u16* row = qkv + tg * (3 * DIM_OUT);
      kv = row[DIM_OUT + h * HD + ch];
      vv = row[2 * DIM_OUT + h * HD + ch];
    }
    size_t base = ((size_t)w * 4 + h) * 4096;
    kp[base + (size_t)tok * 64 + ch] = kv;
    vt[base + (size_t)ch * 64 + tok] = vv;
  }
  // q: pooled 2x2 max, 4 heads * 16 qtok * 64 channels
  for (int idx = threadIdx.x; idx < 4 * 16 * 64; idx += blockDim.x) {
    int h = idx >> 10, qt = (idx >> 6) & 15, ch = idx & 63;
    u16 r = 0;
    if (ch < HD) {
      int qy = qt >> 2, qx = qt & 3;
      float mx = -INFINITY;
#pragma unroll
      for (int dy = 0; dy < 2; ++dy)
#pragma unroll
        for (int dx = 0; dx < 2; ++dx) {
          size_t tg = (size_t)b * 16384 + (size_t)(wy * 8 + qy * 2 + dy) * 128 + (wx * 8 + qx * 2 + dx);
          mx = fmaxf(mx, bf2f(qkv[tg * (3 * DIM_OUT) + h * HD + ch]));
        }
      r = f2bf(mx * ASCALE);
    }
    qp[((size_t)w * 4 + h) * 1024 + (size_t)qt * 64 + ch] = r;
  }
}

// ---------------------------------------------------------------------------
// Attention: one window per block, one wave per head.
// S = q*k^T (WMMA, K=64 padded) -> softmax (16-lane shfl) -> P (LDS, bf16)
// O = P*V   (WMMA, K=64 tokens) -> ao[w*16+t][h*56+ch] bf16
// ---------------------------------------------------------------------------
__global__ __launch_bounds__(128) void attn_kernel(const u16* __restrict__ qp,
                                                   const u16* __restrict__ kp,
                                                   const u16* __restrict__ vt,
                                                   u16* __restrict__ ao) {
  __shared__ u16 pls[4][16][64];
  const int h = threadIdx.x >> 5;
  const int lane = threadIdx.x & 31;
  const int half = lane >> 4;
  const int mrow = lane & 15;
  const int w = blockIdx.x;
  const size_t wh = (size_t)w * 4 + h;

  const u16* qb = qp + wh * 1024;
  const u16* kb = kp + wh * 4096;
  const u16* vb = vt + wh * 4096;

  v8f s0 = {0,0,0,0,0,0,0,0};
  v8f s1 = s0, s2 = s0, s3 = s0;

#pragma unroll
  for (int k0 = 0; k0 < 64; k0 += 32) {
    Frag a, b0, b1, b2, b3;
    a.u[0] = *(const v4u*)(qb + mrow * 64 + k0 + half * 16);
    a.u[1] = *(const v4u*)(qb + mrow * 64 + k0 + half * 16 + 8);
    const v4u* p0 = (const v4u*)(kb + (0 * 16 + mrow) * 64 + k0 + half * 16);
    const v4u* p1 = (const v4u*)(kb + (1 * 16 + mrow) * 64 + k0 + half * 16);
    const v4u* p2 = (const v4u*)(kb + (2 * 16 + mrow) * 64 + k0 + half * 16);
    const v4u* p3 = (const v4u*)(kb + (3 * 16 + mrow) * 64 + k0 + half * 16);
    b0.u[0] = p0[0]; b0.u[1] = p0[1];
    b1.u[0] = p1[0]; b1.u[1] = p1[1];
    b2.u[0] = p2[0]; b2.u[1] = p2[1];
    b3.u[0] = p3[0]; b3.u[1] = p3[1];
    s0 = __builtin_amdgcn_wmma_f32_16x16x32_bf16(false, a.v, false, b0.v, (short)0, s0, false, false);
    s1 = __builtin_amdgcn_wmma_f32_16x16x32_bf16(false, a.v, false, b1.v, (short)0, s1, false, false);
    s2 = __builtin_amdgcn_wmma_f32_16x16x32_bf16(false, a.v, false, b2.v, (short)0, s2, false, false);
    s3 = __builtin_amdgcn_wmma_f32_16x16x32_bf16(false, a.v, false, b3.v, (short)0, s3, false, false);
  }

  // Row-wise softmax. Row m = r + 8*half lives in the 16 lanes of one half.
#pragma unroll
  for (int r = 0; r < 8; ++r) {
    float v0 = s0[r], v1 = s1[r], v2 = s2[r], v3 = s3[r];
    float mx = wredmax16(fmaxf(fmaxf(v0, v1), fmaxf(v2, v3)));
    float e0 = expf(v0 - mx), e1 = expf(v1 - mx), e2 = expf(v2 - mx), e3 = expf(v3 - mx);
    float inv = 1.0f / wredsum16(e0 + e1 + e2 + e3);
    int rr = r + 8 * half;
    pls[h][rr][mrow]      = f2bf(e0 * inv);
    pls[h][rr][16 + mrow] = f2bf(e1 * inv);
    pls[h][rr][32 + mrow] = f2bf(e2 * inv);
    pls[h][rr][48 + mrow] = f2bf(e3 * inv);
  }
  __syncthreads();

  v8f o0 = {0,0,0,0,0,0,0,0};
  v8f o1 = o0, o2 = o0, o3 = o0;
#pragma unroll
  for (int k0 = 0; k0 < 64; k0 += 32) {
    Frag a, b0, b1, b2, b3;
    a.u[0] = *(const v4u*)&pls[h][mrow][k0 + half * 16];
    a.u[1] = *(const v4u*)&pls[h][mrow][k0 + half * 16 + 8];
    const v4u* p0 = (const v4u*)(vb + (0 * 16 + mrow) * 64 + k0 + half * 16);
    const v4u* p1 = (const v4u*)(vb + (1 * 16 + mrow) * 64 + k0 + half * 16);
    const v4u* p2 = (const v4u*)(vb + (2 * 16 + mrow) * 64 + k0 + half * 16);
    const v4u* p3 = (const v4u*)(vb + (3 * 16 + mrow) * 64 + k0 + half * 16);
    b0.u[0] = p0[0]; b0.u[1] = p0[1];
    b1.u[0] = p1[0]; b1.u[1] = p1[1];
    b2.u[0] = p2[0]; b2.u[1] = p2[1];
    b3.u[0] = p3[0]; b3.u[1] = p3[1];
    o0 = __builtin_amdgcn_wmma_f32_16x16x32_bf16(false, a.v, false, b0.v, (short)0, o0, false, false);
    o1 = __builtin_amdgcn_wmma_f32_16x16x32_bf16(false, a.v, false, b1.v, (short)0, o1, false, false);
    o2 = __builtin_amdgcn_wmma_f32_16x16x32_bf16(false, a.v, false, b2.v, (short)0, o2, false, false);
    o3 = __builtin_amdgcn_wmma_f32_16x16x32_bf16(false, a.v, false, b3.v, (short)0, o3, false, false);
  }

  v8f os[4] = {o0, o1, o2, o3};
#pragma unroll
  for (int t = 0; t < 4; ++t) {
    int ch = t * 16 + mrow;
    if (ch >= HD) continue;
#pragma unroll
    for (int r = 0; r < 8; ++r) {
      ao[((size_t)w * 16 + r + 8 * half) * DIM_OUT + h * HD + ch] = f2bf(os[t][r]);
    }
  }
}

// ---------------------------------------------------------------------------
// hs2 = maxpool2(res_proj) + attn_out (spatial order); LN2 -> bf16
// One wave per pooled token (224 ch = 7 per lane).
// ---------------------------------------------------------------------------
__global__ __launch_bounds__(256) void fuse_residual_ln2(const float* __restrict__ rp,
                                                         const float* __restrict__ ob,
                                                         const float* __restrict__ g,
                                                         const float* __restrict__ bt,
                                                         float* __restrict__ hs2,
                                                         u16* __restrict__ ln2) {
  const int wave = threadIdx.x >> 5, lane = threadIdx.x & 31;
  const size_t tok = (size_t)blockIdx.x * 8 + wave;
  const int b = (int)(tok >> 12);
  const int y2 = (int)((tok >> 6) & 63);
  const int x2 = (int)(tok & 63);
  const int wy = y2 >> 2, qy = y2 & 3, wx = x2 >> 2, qx = x2 & 3;
  const size_t orow = (((size_t)(b * 256 + wy * 16 + wx)) * 16 + qy * 4 + qx) * DIM_OUT;
  const size_t r0 = ((size_t)b * 16384 + (size_t)(2 * y2) * 128 + (size_t)(2 * x2)) * DIM_OUT;

  float v[7];
  float s = 0.f, sq = 0.f;
#pragma unroll
  for (int j = 0; j < 7; ++j) {
    int c = lane + 32 * j;
    float m = fmaxf(fmaxf(rp[r0 + c], rp[r0 + DIM_OUT + c]),
                    fmaxf(rp[r0 + 128 * DIM_OUT + c], rp[r0 + 129 * DIM_OUT + c]));
    float val = m + ob[orow + c];
    v[j] = val;
    s += val;
    sq += val * val;
  }
  float mean = wredsum32(s) * (1.0f / DIM_OUT);
  float msq = wredsum32(sq) * (1.0f / DIM_OUT);
  float rstd = rsqrtf(msq - mean * mean + EPSLN);
#pragma unroll
  for (int j = 0; j < 7; ++j) {
    int c = lane + 32 * j;
    hs2[tok * DIM_OUT + c] = v[j];
    ln2[tok * DIM_OUT + c] = f2bf((v[j] - mean) * rstd * g[c] + bt[c]);
  }
}

// ---------------------------------------------------------------------------
// Host orchestration
// ---------------------------------------------------------------------------
extern "C" void kernel_launch(void* const* d_in, const int* in_sizes, int n_in,
                              void* d_out, int out_size, void* d_ws, size_t ws_size,
                              hipStream_t stream) {
  (void)in_sizes; (void)n_in; (void)out_size; (void)ws_size;
  const float* x      = (const float*)d_in[0];
  const float* ln1_g  = (const float*)d_in[1];
  const float* ln1_b  = (const float*)d_in[2];
  const float* qkv_w  = (const float*)d_in[3];
  const float* qkv_b  = (const float*)d_in[4];
  const float* attn_w = (const float*)d_in[5];
  const float* attn_b = (const float*)d_in[6];
  const float* res_w  = (const float*)d_in[7];
  const float* res_b  = (const float*)d_in[8];
  const float* ln2_g  = (const float*)d_in[9];
  const float* ln2_b  = (const float*)d_in[10];
  const float* m1_w   = (const float*)d_in[11];
  const float* m1_b   = (const float*)d_in[12];
  const float* m2_w   = (const float*)d_in[13];
  const float* m2_b   = (const float*)d_in[14];
  float* out = (float*)d_out;

  char* ws = (char*)d_ws;
  size_t off = 0;
  auto alloc = [&](size_t bytes) { size_t o = off; off = (off + bytes + 255) & ~(size_t)255; return o; };

  u16*   hsb    = (u16*)  (ws + alloc(NTOK * 128 * 2));
  u16*   wqkvT  = (u16*)  (ws + alloc(704ull * 128 * 2));
  float* bqkv   = (float*)(ws + alloc(704ull * 4));
  u16*   wresT  = (u16*)  (ws + alloc(256ull * 128 * 2));
  float* bres   = (float*)(ws + alloc(256ull * 4));
  u16*   wattnT = (u16*)  (ws + alloc(256ull * 224 * 2));
  float* battn  = (float*)(ws + alloc(256ull * 4));
  u16*   wm1T   = (u16*)  (ws + alloc(896ull * 224 * 2));
  float* bm1    = (float*)(ws + alloc(896ull * 4));
  u16*   wm2T   = (u16*)  (ws + alloc(256ull * 896 * 2));
  float* bm2    = (float*)(ws + alloc(256ull * 4));
  u16*   qkvb   = (u16*)  (ws + alloc(NTOK * 672 * 2));
  float* rp     = (float*)(ws + alloc(NTOK * 224 * 4));
  u16*   qp     = (u16*)  (ws + alloc((size_t)NWIN * 4 * 16 * 64 * 2));
  u16*   kp     = (u16*)  (ws + alloc((size_t)NWIN * 4 * 64 * 64 * 2));
  u16*   vt     = (u16*)  (ws + alloc((size_t)NWIN * 4 * 64 * 64 * 2));
  u16*   ao     = (u16*)  (ws + alloc(NTOK2 * 224 * 2));
  float* obuf   = (float*)(ws + alloc(NTOK2 * 224 * 4));
  float* hs2    = (float*)(ws + alloc(NTOK2 * 224 * 4));
  u16*   ln2bf  = (u16*)  (ws + alloc(NTOK2 * 224 * 2));
  u16*   m1buf  = (u16*)  (ws + alloc(NTOK2 * 896 * 2));

  // Weight conversion (bf16, transposed, padded)
  convert_weight_t<<<(704 * 128 + 255) / 256, 256, 0, stream>>>(qkv_w,  wqkvT,  112, 672, 128, 704);
  convert_weight_t<<<(256 * 128 + 255) / 256, 256, 0, stream>>>(res_w,  wresT,  112, 224, 128, 256);
  convert_weight_t<<<(256 * 224 + 255) / 256, 256, 0, stream>>>(attn_w, wattnT, 224, 224, 224, 256);
  convert_weight_t<<<(896 * 224 + 255) / 256, 256, 0, stream>>>(m1_w,   wm1T,   224, 896, 224, 896);
  convert_weight_t<<<(256 * 896 + 255) / 256, 256, 0, stream>>>(m2_w,   wm2T,   896, 224, 896, 256);
  pad_bias<<<3, 256, 0, stream>>>(qkv_b,  bqkv,  672, 704);
  pad_bias<<<1, 256, 0, stream>>>(res_b,  bres,  224, 256);
  pad_bias<<<1, 256, 0, stream>>>(attn_b, battn, 224, 256);
  pad_bias<<<4, 256, 0, stream>>>(m1_b,   bm1,   896, 896);
  pad_bias<<<1, 256, 0, stream>>>(m2_b,   bm2,   224, 256);

  // LN1 -> bf16 (padded K=128)
  ln1_to_bf16<<<NTOK / 8, 256, 0, stream>>>(x, ln1_g, ln1_b, hsb);

  // qkv GEMM: [131072 x 128] x [128 x 672] -> bf16
  { dim3 g(11, NTOK / 256);
    gemm_bf16_wmma<<<g, 256, 0, stream>>>(hsb, wqkvT, bqkv, 128, 672, 672, nullptr, qkvb, nullptr, EP_BF16); }

  // res_proj GEMM: [131072 x 128] x [128 x 224] -> f32
  { dim3 g(4, NTOK / 256);
    gemm_bf16_wmma<<<g, 256, 0, stream>>>(hsb, wresT, bres, 128, 224, 224, rp, nullptr, nullptr, EP_F32); }

  // Attention prep + attention
  repack_qkv<<<NWIN, 256, 0, stream>>>(qkvb, qp, kp, vt);
  attn_kernel<<<NWIN, 128, 0, stream>>>(qp, kp, vt, ao);

  // attn_proj GEMM: [32768 x 224] x [224 x 224] -> f32
  { dim3 g(4, NTOK2 / 256);
    gemm_bf16_wmma<<<g, 256, 0, stream>>>(ao, wattnT, battn, 224, 224, 224, obuf, nullptr, nullptr, EP_F32); }

  // residual + attn output, LN2
  fuse_residual_ln2<<<NTOK2 / 8, 256, 0, stream>>>(rp, obuf, ln2_g, ln2_b, hs2, ln2bf);

  // MLP1: [32768 x 224] x [224 x 896] -> GELU -> bf16
  { dim3 g(14, NTOK2 / 256);
    gemm_bf16_wmma<<<g, 256, 0, stream>>>(ln2bf, wm1T, bm1, 224, 896, 896, nullptr, m1buf, nullptr, EP_GELU_BF16); }

  // MLP2: [32768 x 896] x [896 x 224] + hs2 -> out (f32)
  { dim3 g(4, NTOK2 / 256);
    gemm_bf16_wmma<<<g, 256, 0, stream>>>(m1buf, wm2T, bm2, 896, 224, 224, out, nullptr, hs2, EP_ADD_F32); }
}
